// DoubleAttention_11175504904767
// MI455X (gfx1250) — matrix operand order repaired and checked by reading, not compiled, not run on hardware
//
#include <hip/hip_runtime.h>

// Double-Attention (A^2-Net) fused pipeline for gfx1250 (MI455X).
// B=2, C=64, N=32*64*64=131072. Memory-bound: ~406MB traffic @ 23.3TB/s ~ 17.5us.
// Matmuls via v_wmma_f32_16x16x32_f16; softmax stats fused into projection via
// per-wave LDS-transpose reductions; A recomputed on the fly (never hits HBM);
// Y logits stored as packed f16 (row-pairs per dword) with NT hints;
// x / YB / YV tiles staged with global_load_async_to_lds_b32 + s_wait_asynccnt.

typedef __attribute__((ext_vector_type(16))) _Float16 v16h;
typedef __attribute__((ext_vector_type(8)))  float    v8f;

#define B_   2
#define C_   64
#define N_   131072
#define NR   128            // softmax logit rows per batch: [Blogits(64); Vlogits(64)]
#define NCH  (N_ / 128)     // 1024 projection chunks per batch

#if defined(__has_builtin)
#if __has_builtin(__builtin_amdgcn_global_load_async_to_lds_b32) && \
    __has_builtin(__builtin_amdgcn_s_wait_asynccnt)
#define HAVE_ASYNC_LDS 1
#endif
#endif

#if defined(HAVE_ASYNC_LDS)
typedef __attribute__((address_space(1))) int GAS_int;
typedef __attribute__((address_space(3))) int LAS_int;
__device__ inline void async_g2l_b32(const void* g, void* l) {
  __builtin_amdgcn_global_load_async_to_lds_b32((GAS_int*)(g), (LAS_int*)(l), 0, 0);
}
#endif

union PK2 { unsigned u; _Float16 h[2]; };

// ---- WMMA operand loaders (LDS -> frags, layouts per CDNA5 ISA 7.12.2) -----

// A-matrix 16x32 f16: row m = m0 + lane%16; lane<16 holds K 0-7 (v0-3) & 16-23
// (v4-7); lane>=16 holds K 8-15 & 24-31. smem layout: [row][k], k contiguous.
__device__ inline v16h ldsA(const _Float16* s, int m0, int k0, int ld) {
  const int lane = threadIdx.x & 31;
  const _Float16* p = s + (size_t)(m0 + (lane & 15)) * ld + k0 + ((lane >> 4) << 3);
  union { v16h v; unsigned u[8]; } r;
#pragma unroll
  for (int i = 0; i < 8; ++i) {
    const int off = ((i & 4) << 2) + ((i & 3) << 1);  // {0,2,4,6,16,18,20,22}
    r.u[i] = *(const unsigned*)(p + off);
  }
  return r.v;
}

// B-matrix 32x16 f16: column n = n0 + lane%16; lanes 0-15 hold K 0-7 & 16-23,
// lanes 16-31 hold K 8-15 & 24-31. smem layout: [col][k], k contiguous.
__device__ inline v16h ldsB(const _Float16* s, int n0, int k0, int ld) {
  const int lane = threadIdx.x & 31;
  const _Float16* p = s + (size_t)(n0 + (lane & 15)) * ld + k0 + ((lane >> 4) << 3);
  union { v16h v; unsigned u[8]; } r;
#pragma unroll
  for (int i = 0; i < 8; ++i) r.u[i] = *(const unsigned*)(p + ((i & 4) << 2) + ((i & 3) << 1));
  return r.v;
}

__device__ inline v8f wmma16(v16h a, v16h b, v8f c) {
  return __builtin_amdgcn_wmma_f32_16x16x32_f16(false, a, false, b, (short)0, c,
                                                false, false);
}

__device__ inline float max4(float4 q) {
  return fmaxf(fmaxf(q.x, q.y), fmaxf(q.z, q.w));
}
__device__ inline float sumexp4(float4 q, float m) {
  return __expf(q.x - m) + __expf(q.y - m) + __expf(q.z - m) + __expf(q.w - m);
}

// ---- K0: zero attn_map accumulator -----------------------------------------
__global__ __launch_bounds__(256) void zero_kernel(float* p, int n) {
  int i = blockIdx.x * 256 + threadIdx.x;
  if (i < n) p[i] = 0.f;
}

// ---- K1: Ypk[b,q,n] = packed f16 rows (2q,2q+1) of [Wb;Wv]@x + bias;
//          fused per-chunk softmax (max,sumexp) partials from f32 accumulators.
__global__ __launch_bounds__(256) void proj_kernel(
    const float* __restrict__ x,
    const float* __restrict__ Wb, const float* __restrict__ bb,
    const float* __restrict__ Wv, const float* __restrict__ bv,
    unsigned* __restrict__ Ypk, float* __restrict__ pmax, float* __restrict__ psum) {
  __shared__ __align__(16) _Float16 Wl[NR * 72];
  __shared__ __align__(16) _Float16 Xl[128 * 72];
  __shared__ float bl[NR];
  __shared__ float rmax[NR * 8], rsum[NR * 8];
  __shared__ __align__(16) float Tb[8 * 16 * 20];   // per-wave transpose tiles
#if defined(HAVE_ASYNC_LDS)
  __shared__ __align__(16) float Raw[64 * 128];
#endif
  const int tid   = threadIdx.x;
  const int b     = blockIdx.x / NCH;
  const int chunk = blockIdx.x % NCH;
  const int n0    = chunk * 128;
  const float* xb = x + (size_t)b * C_ * N_;

#if defined(HAVE_ASYNC_LDS)
  for (int i = tid; i < 64 * 128; i += 256) {        // async raw x -> LDS first
    const int c = i >> 7, n = i & 127;
    async_g2l_b32(&xb[(size_t)c * N_ + n0 + n], &Raw[c * 128 + n]);
  }
#endif
  for (int i = tid; i < 64 * 64; i += 256) {         // stage weights (f16)
    const int m = i >> 6, k = i & 63;
    Wl[(size_t)m * 72 + k]        = (_Float16)Wb[i];
    Wl[(size_t)(m + 64) * 72 + k] = (_Float16)Wv[i];
  }
  if (tid < NR) bl[tid] = (tid < 64) ? bb[tid] : bv[tid - 64];
#if defined(HAVE_ASYNC_LDS)
  __builtin_amdgcn_s_wait_asynccnt(0);
  __syncthreads();
  for (int i = tid; i < 64 * 128; i += 256) {        // convert f32 -> f16 (transposed)
    const int c = i >> 7, n = i & 127;
    Xl[(size_t)n * 72 + c] = (_Float16)Raw[c * 128 + n];
  }
#else
  for (int i = tid; i < 64 * 128; i += 256) {
    const int c = i >> 7, n = i & 127;
    Xl[(size_t)n * 72 + c] = (_Float16)xb[(size_t)c * N_ + n0 + n];
  }
#endif
  __syncthreads();

  const int wave = tid >> 5, lane = tid & 31;
  const v16h bx0 = ldsB(Xl, wave * 16, 0, 72);
  const v16h bx1 = ldsB(Xl, wave * 16, 32, 72);
  unsigned* Yb = Ypk + (size_t)b * 64 * N_;          // 64 packed row-pairs
  float* T = Tb + wave * (16 * 20);
  const int n = n0 + wave * 16 + (lane & 15);
#pragma unroll 1
  for (int mt = 0; mt < 8; ++mt) {
    const int mrow = mt * 16 + ((lane >> 4) << 3);
    v8f acc;
#pragma unroll
    for (int r = 0; r < 8; ++r) acc[r] = bl[mrow + r];
    acc = wmma16(ldsA(Wl, mt * 16, 0, 72), bx0, acc);
    acc = wmma16(ldsA(Wl, mt * 16, 32, 72), bx1, acc);
#pragma unroll
    for (int j = 0; j < 4; ++j) {                    // packed f16 row-pair stores
      PK2 pk;
      pk.h[0] = (_Float16)acc[2 * j];
      pk.h[1] = (_Float16)acc[2 * j + 1];
      __builtin_nontemporal_store(pk.u, &Yb[(size_t)((mrow >> 1) + j) * N_ + n]);
    }
#pragma unroll
    for (int r = 0; r < 8; ++r)                      // wave-private transpose tile
      T[(((lane >> 4) << 3) + r) * 20 + (lane & 15)] = acc[r];
    // lanes 0-15: reduce one full row each (LDS in-order within wave, no barrier)
    if (lane < 16) {
      const float* row = T + lane * 20;
      const float4 q0 = *(const float4*)(row);
      const float4 q1 = *(const float4*)(row + 4);
      const float4 q2 = *(const float4*)(row + 8);
      const float4 q3 = *(const float4*)(row + 12);
      const float m = fmaxf(fmaxf(max4(q0), max4(q1)), fmaxf(max4(q2), max4(q3)));
      const float s = sumexp4(q0, m) + sumexp4(q1, m) + sumexp4(q2, m) + sumexp4(q3, m);
      rmax[(mt * 16 + lane) * 8 + wave] = m;
      rsum[(mt * 16 + lane) * 8 + wave] = s;
    }
  }
  __syncthreads();
  if (tid < NR) {                         // combine 8 wave partials per row
    float M = -3.0e38f, S = 0.f;
#pragma unroll
    for (int w = 0; w < 8; ++w) {
      const float m2 = rmax[tid * 8 + w], s2 = rsum[tid * 8 + w];
      const float Mn = fmaxf(M, m2);
      S = S * __expf(M - Mn) + s2 * __expf(m2 - Mn);
      M = Mn;
    }
    pmax[(size_t)(b * NR + tid) * NCH + chunk] = M;
    psum[(size_t)(b * NR + tid) * NCH + chunk] = S;
  }
}

// ---- K2: reduce per-chunk partials -> stats[0:256)=max, [256:512)=sumexp ---
__global__ __launch_bounds__(256) void stats_kernel(const float* __restrict__ pmax,
                                                    const float* __restrict__ psum,
                                                    float* __restrict__ stats) {
  const int row = blockIdx.x;             // 0 .. B*NR-1
  const float* pm = pmax + (size_t)row * NCH;
  const float* ps = psum + (size_t)row * NCH;
  float M = -3.0e38f, S = 0.f;
  for (int i = threadIdx.x; i < NCH; i += 256) {
    const float m2 = pm[i], s2 = ps[i];
    const float Mn = fmaxf(M, m2);
    S = S * __expf(M - Mn) + s2 * __expf(m2 - Mn);
    M = Mn;
  }
  __shared__ float sm[256], ss[256];
  sm[threadIdx.x] = M; ss[threadIdx.x] = S;
  __syncthreads();
  for (int st = 128; st > 0; st >>= 1) {
    if (threadIdx.x < st) {
      const float m1 = sm[threadIdx.x], s1 = ss[threadIdx.x];
      const float m2 = sm[threadIdx.x + st], s2 = ss[threadIdx.x + st];
      const float Mx = fmaxf(m1, m2);
      sm[threadIdx.x] = Mx;
      ss[threadIdx.x] = s1 * __expf(m1 - Mx) + s2 * __expf(m2 - Mx);
    }
    __syncthreads();
  }
  if (threadIdx.x == 0) { stats[row] = sm[0]; stats[256 + row] = ss[0]; }
}

// ---- K3: recompute A = Wa@x + ba on the fly; attn_map += A @ expB^T --------
__global__ __launch_bounds__(256) void attnmap_kernel(
    const float* __restrict__ x, const float* __restrict__ Wa,
    const float* __restrict__ ba, const unsigned* __restrict__ Ypk,
    const float* __restrict__ stats, float* __restrict__ am) {
  __shared__ __align__(16) _Float16 Wal[64 * 72];
  __shared__ __align__(16) _Float16 Xl[128 * 72];
  __shared__ __align__(16) _Float16 Al[64 * 136];
  __shared__ __align__(16) _Float16 Bl[64 * 136];
  __shared__ float bal[64];
#if defined(HAVE_ASYNC_LDS)
  __shared__ __align__(16) unsigned RawU[32 * 128];
#endif
  const int tid = threadIdx.x;
  const int b   = blockIdx.x / (N_ / 512);
  const int n0  = (blockIdx.x % (N_ / 512)) * 512;
  for (int i = tid; i < 64 * 64; i += 256) {
    const int m = i >> 6, k = i & 63;
    Wal[(size_t)m * 72 + k] = (_Float16)Wa[i];
  }
  if (tid < 64) bal[tid] = ba[tid];
  const float* xb = x + (size_t)b * C_ * N_;
  const unsigned* YB = Ypk + (size_t)b * 64 * N_;    // B rows: pairs q = 0..31
  const int wave = tid >> 5, lane = tid & 31;
  const int mt_c = wave >> 1;
  const int kt0  = (wave & 1) * 2;
  v8f acc0 = {}, acc1 = {};
#pragma unroll 1
  for (int blk = 0; blk < 4; ++blk) {
    const int nb = n0 + blk * 128;
    __syncthreads();                                   // protect prior tiles
#if defined(HAVE_ASYNC_LDS)
    for (int i = tid; i < 32 * 128; i += 256) {        // async packed YB -> LDS
      const int q = i >> 7, n = i & 127;
      async_g2l_b32(&YB[(size_t)q * N_ + nb + n], &RawU[q * 128 + n]);
    }
#endif
    for (int i = tid; i < 64 * 128; i += 256) {        // stage x transposed
      const int c = i >> 7, n = i & 127;
      Xl[(size_t)n * 72 + c] = (_Float16)xb[(size_t)c * N_ + nb + n];
    }
#if defined(HAVE_ASYNC_LDS)
    __builtin_amdgcn_s_wait_asynccnt(0);
    __syncthreads();
    for (int i = tid; i < 32 * 128; i += 256) {        // unpack + exp
      const int q = i >> 7, n = i & 127;
      PK2 pk; pk.u = RawU[q * 128 + n];
      const int k0 = 2 * q;
      Bl[(size_t)k0 * 136 + n] =
          (_Float16)__expf((float)pk.h[0] - stats[b * NR + k0]);
      Bl[(size_t)(k0 + 1) * 136 + n] =
          (_Float16)__expf((float)pk.h[1] - stats[b * NR + k0 + 1]);
    }
#else
    for (int i = tid; i < 32 * 128; i += 256) {
      const int q = i >> 7, n = i & 127;
      PK2 pk; pk.u = __builtin_nontemporal_load(&YB[(size_t)q * N_ + nb + n]);
      const int k0 = 2 * q;
      Bl[(size_t)k0 * 136 + n] =
          (_Float16)__expf((float)pk.h[0] - stats[b * NR + k0]);
      Bl[(size_t)(k0 + 1) * 136 + n] =
          (_Float16)__expf((float)pk.h[1] - stats[b * NR + k0 + 1]);
    }
#endif
    __syncthreads();
    // recompute A tile (Wa @ x + ba) for this wave's 16-column strip
    const v16h fx0 = ldsB(Xl, wave * 16, 0, 72);
    const v16h fx1 = ldsB(Xl, wave * 16, 32, 72);
#pragma unroll
    for (int mt = 0; mt < 4; ++mt) {
      const int mrow = mt * 16 + ((lane >> 4) << 3);
      v8f a;
#pragma unroll
      for (int r = 0; r < 8; ++r) a[r] = bal[mrow + r];
      a = wmma16(ldsA(Wal, mt * 16, 0, 72), fx0, a);
      a = wmma16(ldsA(Wal, mt * 16, 32, 72), fx1, a);
#pragma unroll
      for (int r = 0; r < 8; ++r)
        Al[(size_t)(mrow + r) * 136 + wave * 16 + (lane & 15)] = (_Float16)a[r];
    }
    __syncthreads();
#pragma unroll
    for (int kc = 0; kc < 4; ++kc) {                   // contract over n
      const v16h af = ldsA(Al, mt_c * 16, kc * 32, 136);
      acc0 = wmma16(af, ldsB(Bl, kt0 * 16, kc * 32, 136), acc0);
      acc1 = wmma16(af, ldsB(Bl, (kt0 + 1) * 16, kc * 32, 136), acc1);
    }
  }
  float* amb = am + b * 64 * 64;
  const int mrow = mt_c * 16 + ((lane >> 4) << 3);
  const int kcol = kt0 * 16 + (lane & 15);
#pragma unroll
  for (int r = 0; r < 8; ++r) {
    atomicAdd(&amb[(mrow + r) * 64 + kcol], acc0[r]);
    atomicAdd(&amb[(mrow + r) * 64 + kcol + 16], acc1[r]);
  }
}

// ---- K4: M2[c,k] = (1/sumB_k)(1/sumV_k) * sum_m Wc[c,m] * attn_map[m,k] ----
__global__ __launch_bounds__(256) void combine_kernel(const float* __restrict__ Wc,
                                                      const float* __restrict__ am,
                                                      const float* __restrict__ stats,
                                                      float* __restrict__ M2) {
  const int b = blockIdx.x;
  __shared__ float aml[64 * 64];
  for (int i = threadIdx.x; i < 64 * 64; i += 256) aml[i] = am[b * 4096 + i];
  __syncthreads();
  for (int i = threadIdx.x; i < 64 * 64; i += 256) {
    const int c = i >> 6, k = i & 63;
    float acc = 0.f;
#pragma unroll 4
    for (int m = 0; m < 64; ++m) acc += Wc[c * 64 + m] * aml[m * 64 + k];
    const float invB = 1.f / stats[256 + b * NR + k];
    const float invV = 1.f / stats[256 + b * NR + 64 + k];
    M2[b * 4096 + i] = acc * invB * invV;
  }
}

// ---- K5: out = x + bc + M2 @ exp(YV - maxV) --------------------------------
__global__ __launch_bounds__(256) void out_kernel(const float* __restrict__ x,
                                                  const unsigned* __restrict__ Ypk,
                                                  const float* __restrict__ stats,
                                                  const float* __restrict__ M2,
                                                  const float* __restrict__ bc,
                                                  float* __restrict__ out) {
  __shared__ __align__(16) _Float16 Ml[64 * 72];
  __shared__ __align__(16) _Float16 Vl[128 * 72];
  __shared__ float bl[64];
#if defined(HAVE_ASYNC_LDS)
  __shared__ __align__(16) unsigned RawU[32 * 128];
#endif
  const int tid = threadIdx.x;
  const int b   = blockIdx.x / NCH;
  const int n0  = (blockIdx.x % NCH) * 128;
  // V rows 64..127 are packed pairs q = 32..63
  const unsigned* YV = Ypk + (size_t)b * 64 * N_ + (size_t)32 * N_;
#if defined(HAVE_ASYNC_LDS)
  for (int i = tid; i < 32 * 128; i += 256) {        // async packed YV -> LDS first
    const int q = i >> 7, n = i & 127;
    async_g2l_b32(&YV[(size_t)q * N_ + n0 + n], &RawU[q * 128 + n]);
  }
#endif
  for (int i = tid; i < 64 * 64; i += 256) {
    const int c = i >> 6, k = i & 63;
    Ml[(size_t)c * 72 + k] = (_Float16)M2[b * 4096 + i];
  }
  if (tid < 64) bl[tid] = bc[tid];
#if defined(HAVE_ASYNC_LDS)
  __builtin_amdgcn_s_wait_asynccnt(0);
  __syncthreads();
  for (int i = tid; i < 32 * 128; i += 256) {        // unpack + exp (transposed)
    const int q = i >> 7, n = i & 127;
    PK2 pk; pk.u = RawU[q * 128 + n];
    const int k0 = 2 * q;
    Vl[(size_t)n * 72 + k0] =
        (_Float16)__expf((float)pk.h[0] - stats[b * NR + 64 + k0]);
    Vl[(size_t)n * 72 + k0 + 1] =
        (_Float16)__expf((float)pk.h[1] - stats[b * NR + 64 + k0 + 1]);
  }
#else
  for (int i = tid; i < 32 * 128; i += 256) {
    const int q = i >> 7, n = i & 127;
    PK2 pk; pk.u = __builtin_nontemporal_load(&YV[(size_t)q * N_ + n0 + n]);
    const int k0 = 2 * q;
    Vl[(size_t)n * 72 + k0] =
        (_Float16)__expf((float)pk.h[0] - stats[b * NR + 64 + k0]);
    Vl[(size_t)n * 72 + k0 + 1] =
        (_Float16)__expf((float)pk.h[1] - stats[b * NR + 64 + k0 + 1]);
  }
#endif
  __syncthreads();
  const int wave = tid >> 5, lane = tid & 31;
  const v16h v0 = ldsB(Vl, wave * 16, 0, 72);
  const v16h v1 = ldsB(Vl, wave * 16, 32, 72);
  const float* xb = x + (size_t)b * C_ * N_;
  float* ob = out + (size_t)b * C_ * N_;
  const int n = n0 + wave * 16 + (lane & 15);
#pragma unroll 1
  for (int mt = 0; mt < 4; ++mt) {
    const int mrow = mt * 16 + ((lane >> 4) << 3);
    v8f acc;
#pragma unroll
    for (int r = 0; r < 8; ++r) acc[r] = bl[mrow + r];
    acc = wmma16(ldsA(Ml, mt * 16, 0, 72), v0, acc);
    acc = wmma16(ldsA(Ml, mt * 16, 32, 72), v1, acc);
#pragma unroll
    for (int r = 0; r < 8; ++r) {
      const float xi = __builtin_nontemporal_load(&xb[(size_t)(mrow + r) * N_ + n]);
      __builtin_nontemporal_store(acc[r] + xi, &ob[(size_t)(mrow + r) * N_ + n]);
    }
  }
}

extern "C" void kernel_launch(void* const* d_in, const int* in_sizes, int n_in,
                              void* d_out, int out_size, void* d_ws, size_t ws_size,
                              hipStream_t stream) {
  const float* x  = (const float*)d_in[0];
  const float* Wa = (const float*)d_in[1];
  const float* ba = (const float*)d_in[2];
  const float* Wb = (const float*)d_in[3];
  const float* bb = (const float*)d_in[4];
  const float* Wv = (const float*)d_in[5];
  const float* bv = (const float*)d_in[6];
  const float* Wc = (const float*)d_in[7];
  const float* bc = (const float*)d_in[8];
  float* out = (float*)d_out;

  float* ws      = (float*)d_ws;
  unsigned* Ypk  = (unsigned*)ws;                    // B*64*N dwords (~67MB packed f16)
  float* pmax    = ws + (size_t)B_ * 64 * N_;        // B*128*NCH partial maxes
  float* psum    = pmax + (size_t)B_ * NR * NCH;     // B*128*NCH partial sums
  float* stats   = psum + (size_t)B_ * NR * NCH;     // 512: max | sumexp
  float* am      = stats + 512;                      // B*64*64 attn_map (raw)
  float* M2      = am + (size_t)B_ * 4096;           // B*64*64 fused out-proj

  zero_kernel<<<(B_ * 4096 + 255) / 256, 256, 0, stream>>>(am, B_ * 4096);
  proj_kernel<<<B_ * NCH, 256, 0, stream>>>(x, Wb, bb, Wv, bv, Ypk, pmax, psum);
  stats_kernel<<<B_ * NR, 256, 0, stream>>>(pmax, psum, stats);
  attnmap_kernel<<<B_ * (N_ / 512), 256, 0, stream>>>(x, Wa, ba, Ypk, stats, am);
  combine_kernel<<<B_, 256, 0, stream>>>(Wc, am, stats, M2);
  out_kernel<<<B_ * NCH, 256, 0, stream>>>(x, Ypk, stats, M2, bc, out);
}
